// DCNv3C2f_10840497455263
// MI455X (gfx1250) — matrix operand order repaired and checked by reading, not compiled, not run on hardware
//
#include <hip/hip_runtime.h>
#include <hip/hip_bf16.h>

// ---------------------------------------------------------------------------
// C2f + BottleneckDCNv3 pipeline for MI455X (gfx1250, wave32, WMMA + TDM).
// Conv GEMMs: v_wmma_f32_16x16x32_bf16, 64x128 macro-tile, 4 WMMA / wave / K.
// Weight tiles staged into LDS by the Tensor Data Mover (tensor_load_to_lds,
// TENSORcnt). DCN sampling / depthwise / softmax are VALU+VMEM kernels.
// ---------------------------------------------------------------------------

typedef __attribute__((ext_vector_type(16))) __bf16 v16bf;
typedef __attribute__((ext_vector_type(8)))  float  v8f;

// TDM descriptor group types (per CDNA5_HIP.md probe)
typedef __attribute__((ext_vector_type(4))) unsigned int tdm_v4u;
typedef __attribute__((ext_vector_type(8))) int          tdm_v8i;
typedef __attribute__((ext_vector_type(4))) int          tdm_v4i;

#if defined(__has_builtin)
#if __has_builtin(__builtin_amdgcn_tensor_load_to_lds) && \
    __has_builtin(__builtin_amdgcn_s_wait_tensorcnt)
#define USE_TDM 1
#endif
#endif
#ifndef USE_TDM
#define USE_TDM 0
#endif

union FragU {
    unsigned int u[8];
    v16bf        v;
};

__device__ __forceinline__ unsigned short f2bf(float f) {
    union { float f; unsigned int u; } c; c.f = f;
    unsigned int r = c.u + 0x7FFFu + ((c.u >> 16) & 1u);   // round-to-nearest-even
    return (unsigned short)(r >> 16);
}
__device__ __forceinline__ float bf2f(unsigned short h) {
    union { unsigned int u; float f; } c; c.u = ((unsigned int)h) << 16;
    return c.f;
}
__device__ __forceinline__ float silu(float v) {
    return v / (1.0f + __expf(-v));
}

// ---------------------------------------------------------------------------
// fp32 -> bf16 bulk conversion
// ---------------------------------------------------------------------------
__global__ void f32_to_bf16_kernel(const float* __restrict__ src,
                                   unsigned short* __restrict__ dst, long n) {
    long i = (long)blockIdx.x * blockDim.x + threadIdx.x;
    if (i < n) dst[i] = f2bf(src[i]);
}

// ---------------------------------------------------------------------------
// Generic conv-as-GEMM with WMMA bf16 + TDM-staged weight tiles.
//   Out[m, n] = act(scale[m] * sum_k A[m,k] * B[k,n] + bias[m])
//   B[k,n] gathered from NCHW input: k=(ci,kh,kw) for kk==3, k=ci for kk==1,
//   n=(b,hw). Macro tile 64(M) x 128(N), 256 threads = 8 waves (4Mx2N),
//   each wave: 16x64 strip of C = 4 accumulators, 4 WMMAs per K-step of 32.
// ---------------------------------------------------------------------------
#define TILE_M 64
#define TILE_N 128
#define TILE_K 32

__global__ __launch_bounds__(256)
void gemm_conv_wmma(const unsigned short* __restrict__ A,   // bf16 (M,K)
                    const unsigned short* __restrict__ In,  // bf16 (B,inCtot,HW)
                    const float* __restrict__ scale,        // nullable (=> 1.0)
                    const float* __restrict__ bias,
                    float* __restrict__ outF,               // nullable
                    unsigned short* __restrict__ outB,      // nullable
                    int M, int K, int kk, int HW, int Wimg,
                    int inCtot, int inChOff, int outCtot, int outChOff,
                    int act) {
    __shared__ unsigned short ldsA[TILE_M * TILE_K];   // [m][k]   4 KB
    __shared__ unsigned short ldsB[TILE_N * TILE_K];   // [n][k]   8 KB (transposed)

    const int tid  = threadIdx.x;
    const int lane = tid & 31;
    const int wave = tid >> 5;
    const int wm = wave >> 1;          // 0..3  (M sub-tile, 16 rows)
    const int wn = wave & 1;           // 0..1  (N half, 64 cols)

    const int tilesPerImg = HW / TILE_N;               // 50
    const int b   = blockIdx.x / tilesPerImg;
    const int hw0 = (blockIdx.x % tilesPerImg) * TILE_N;
    const int m0  = blockIdx.y * TILE_M;
    const int Himg = HW / Wimg;

    const long inBase = ((long)b * inCtot + inChOff) * (long)HW;

#if !USE_TDM
    // A staging (fallback): 64 rows x 32 halves, 8 halves (uint4) per thread
    const int aRow = tid >> 2;
    const int aSeg = (tid & 3) * 8;
#endif

    // B staging (kk==3): 16 elements per thread, n fixed per thread
    const int nl3  = tid >> 1;                 // 0..127
    const int kl3  = (tid & 1) * 16;           // 0 or 16
    const int hw3  = hw0 + nl3;
    const int h3   = hw3 / Wimg;
    const int w3   = hw3 - h3 * Wimg;

    v8f acc0 = {}, acc1 = {}, acc2 = {}, acc3 = {};
    const int nIter = K / TILE_K;

    for (int kt = 0; kt < nIter; ++kt) {
        const int k0 = kt * TILE_K;

        // ---- stage A tile (weights, 64 x 32 bf16) ----
#if USE_TDM
        if (tid < 32) {   // one wave issues the TDM descriptor
            const unsigned ldsOff = (unsigned)(unsigned long long)(const void*)ldsA;
            const unsigned long long ga =
                (unsigned long long)(const char*)(A + (long)m0 * K + k0);
            int rowsLeft = M - m0; if (rowsLeft < 0) rowsLeft = 0;
            const int colsLeft = K - k0;

            tdm_v4u g0;
            g0.x = 1u;                                   // count=1, user mode
            g0.y = ldsOff;                               // lds_addr
            g0.z = (unsigned)(ga & 0xFFFFFFFFull);       // global_addr[31:0]
            g0.w = (unsigned)((ga >> 32) & 0x01FFFFFFull) | (2u << 30); // type=2

            tdm_v8i g1;
            g1[0] = (int)(1u << 16);                     // data_size = 2 bytes
            g1[1] = (int)(((unsigned)colsLeft & 0xFFFFu) << 16);        // dim0 lo16
            g1[2] = (int)((((unsigned)colsLeft >> 16) & 0xFFFFu) |
                          (((unsigned)rowsLeft & 0xFFFFu) << 16));      // dim0 hi / dim1 lo
            g1[3] = (int)((((unsigned)rowsLeft >> 16) & 0xFFFFu) |
                          ((unsigned)TILE_K << 16));                    // dim1 hi / tile_dim0
            g1[4] = TILE_M;                              // tile_dim1=64, tile_dim2=0
            g1[5] = K;                                   // tensor_dim0_stride lo32
            g1[6] = 0;
            g1[7] = 0;

            tdm_v4i gz = {0, 0, 0, 0};
#if __clang_major__ >= 23
            tdm_v8i gz8 = {0, 0, 0, 0, 0, 0, 0, 0};
            __builtin_amdgcn_tensor_load_to_lds(g0, g1, gz, gz, gz8, 0);
#else
            __builtin_amdgcn_tensor_load_to_lds(g0, g1, gz, gz, 0);
#endif
        }
#else
        {
            uint4 av = make_uint4(0u, 0u, 0u, 0u);
            const int mg = m0 + aRow;
            if (mg < M) {
                av = *(const uint4*)(A + (long)mg * K + k0 + aSeg);
                if (kt + 1 < nIter)
                    __builtin_prefetch(A + (long)mg * K + k0 + TILE_K + aSeg, 0, 1);
            }
            *(uint4*)(&ldsA[aRow * TILE_K + aSeg]) = av;
        }
#endif

        // ---- stage B tile (transposed [n][k] in LDS) ----
        if (kk == 1) {
            // 128n x 32k: 512 uint4 loads (8 contiguous hw for fixed k),
            // 2 per thread.
#pragma unroll
            for (int it = 0; it < 2; ++it) {
                const int t2  = tid + it * 256;          // 0..511
                const int kL  = t2 & 31;
                const int n0  = (t2 >> 5) * 8;           // 0..120
                const uint4 bv = *(const uint4*)(In + inBase +
                                   (long)(k0 + kL) * HW + hw0 + n0);
                const unsigned short* hv = (const unsigned short*)&bv;
#pragma unroll
                for (int i = 0; i < 8; ++i)
                    ldsB[(n0 + i) * TILE_K + kL] = hv[i];
            }
        } else { // 3x3, on-the-fly im2col, pad=1: 16 scalar gathers per thread
#pragma unroll
            for (int i = 0; i < 16; ++i) {
                const int kg = k0 + kl3 + i;
                const int ci = kg / 9;
                const int r  = kg - ci * 9;
                const int kh = r / 3;
                const int kw = r - kh * 3;
                const int hh = h3 + kh - 1;
                const int ww = w3 + kw - 1;
                unsigned short v = 0;
                if (hh >= 0 && hh < Himg && ww >= 0 && ww < Wimg)
                    v = In[inBase + (long)ci * HW + hh * Wimg + ww];
                ldsB[nl3 * TILE_K + kl3 + i] = v;
            }
        }

#if USE_TDM
        if (tid < 32) __builtin_amdgcn_s_wait_tensorcnt(0);
#endif
        __syncthreads();

        // ---- build fragments per ISA VGPR layouts ----
        const unsigned int* la = (const unsigned int*)ldsA;
        const unsigned int* lb = (const unsigned int*)ldsB;

        // A 16x32 bf16: lane m=lane%16; lanes>=16 take K-halves offset by 8.
        FragU fa;
        const int mrow  = wm * 16 + (lane & 15);
        const int khalf = (lane >> 4) * 8;
        const int ab0 = (mrow * TILE_K + khalf) >> 1;
        const int ab1 = (mrow * TILE_K + 16 + khalf) >> 1;
#pragma unroll
        for (int j = 0; j < 4; ++j) fa.u[j]     = la[ab0 + j];
#pragma unroll
        for (int j = 0; j < 4; ++j) fa.u[4 + j] = la[ab1 + j];

        // B 32x16 bf16: lane n=lane%16; lanes>=16 take K-halves offset by 8 (same as A).
        const int kgrp  = (lane >> 4) * 8;
        const int nbase = wn * 64 + (lane & 15);
#pragma unroll
        for (int s = 0; s < 4; ++s) {
            FragU fb;
            const int bb0 = ((nbase + s * 16) * TILE_K + kgrp) >> 1;
#pragma unroll
            for (int j = 0; j < 4; ++j) { fb.u[j] = lb[bb0 + j]; fb.u[4 + j] = lb[bb0 + 8 + j]; }
            v8f& acc = (s == 0) ? acc0 : (s == 1) ? acc1 : (s == 2) ? acc2 : acc3;
            acc = __builtin_amdgcn_wmma_f32_16x16x32_bf16(
                      false, fa.v, false, fb.v, (short)0, acc, false, false);
        }

        __syncthreads();
    }

    // ---- epilogue: scale/bias + SiLU, scatter to NCHW ----
    const int mb = m0 + wm * 16 + ((lane >> 4) << 3);
    const long outBase = ((long)b * outCtot + outChOff) * (long)HW;
#pragma unroll
    for (int s = 0; s < 4; ++s) {
        const v8f& acc = (s == 0) ? acc0 : (s == 1) ? acc1 : (s == 2) ? acc2 : acc3;
        const int nimg = hw0 + wn * 64 + s * 16 + (lane & 15);
#pragma unroll
        for (int r = 0; r < 8; ++r) {
            const int m = mb + r;
            if (m < M) {
                const float sc = scale ? scale[m] : 1.0f;
                float v = acc[r] * sc + bias[m];
                if (act) v = silu(v);
                const long o = outBase + (long)m * HW + nimg;
                if (outF) outF[o] = v;
                if (outB) outB[o] = f2bf(v);
            }
        }
    }
}

// ---------------------------------------------------------------------------
// Depthwise 3x3 conv + BN + SiLU (bf16 in / bf16 out)
// ---------------------------------------------------------------------------
__global__ __launch_bounds__(256)
void dwconv_silu_kernel(const unsigned short* __restrict__ T,
                        const float* __restrict__ Wd,  // (C,1,3,3) fp32
                        const float* __restrict__ s,
                        const float* __restrict__ bb,
                        unsigned short* __restrict__ out,
                        int Bn, int C, int Himg, int Wimg) {
    const int HW = Himg * Wimg;
    long idx = (long)blockIdx.x * blockDim.x + threadIdx.x;
    const long total = (long)Bn * C * HW;
    if (idx >= total) return;
    const int hw = (int)(idx % HW);
    const long bc = idx / HW;
    const int c = (int)(bc % C);
    const int h = hw / Wimg, w = hw - h * Wimg;
    const unsigned short* tp = T + bc * HW;
    const float* wp = Wd + c * 9;
    float accv = 0.f;
#pragma unroll
    for (int kh = 0; kh < 3; ++kh) {
#pragma unroll
        for (int kw = 0; kw < 3; ++kw) {
            const int hh = h + kh - 1, ww = w + kw - 1;
            if (hh >= 0 && hh < Himg && ww >= 0 && ww < Wimg)
                accv += bf2f(tp[hh * Wimg + ww]) * wp[kh * 3 + kw];
        }
    }
    float v = accv * s[c] + bb[c];
    out[idx] = f2bf(silu(v));
}

// ---------------------------------------------------------------------------
// DCNv3 core: softmax over P=9 + bilinear sampling, one thread per (b,g,h,w),
// inner loop over the 32 group channels. Writes bf16 into the concat buffer.
// Coordinate math matches the reference: ix_pad = w + kx + off_x (p = kx*3+ky),
// zero padding ring of width 1 around the 80x80 image.
// ---------------------------------------------------------------------------
__global__ __launch_bounds__(128)
void dcn_core_kernel(const unsigned short* __restrict__ T,   // (B,128,HW) bf16
                     const float* __restrict__ OFF,          // (B,72,HW)
                     const float* __restrict__ MLG,          // (B,36,HW)
                     unsigned short* __restrict__ Ycat,      // (B,512,HW)
                     int outChOff, int Bn, int Himg, int Wimg) {
    const int HW = Himg * Wimg;
    long idx = (long)blockIdx.x * blockDim.x + threadIdx.x;
    const long total = (long)Bn * 4 * HW;
    if (idx >= total) return;
    const int hw = (int)(idx % HW);
    const long t = idx / HW;
    const int g = (int)(t & 3);
    const int b = (int)(t >> 2);
    const int h = hw / Wimg, w = hw - h * Wimg;

    const float* offp = OFF + ((long)b * 72 + g * 18) * HW + hw;
    const float* mlp  = MLG + ((long)b * 36 + g * 9) * HW + hw;

    // softmax over P
    float lg[9];
    float mx = -1e30f;
#pragma unroll
    for (int p = 0; p < 9; ++p) { lg[p] = mlp[(long)p * HW]; mx = fmaxf(mx, lg[p]); }
    float sum = 0.f;
#pragma unroll
    for (int p = 0; p < 9; ++p) { lg[p] = __expf(lg[p] - mx); sum += lg[p]; }
    const float inv = 1.0f / sum;

    // precompute 4 corners per sample point, mask folded into weights
    int   cidx[9][4];
    float cw[9][4];
#pragma unroll
    for (int p = 0; p < 9; ++p) {
        const float ox = offp[(long)(2 * p) * HW];
        const float oy = offp[(long)(2 * p + 1) * HW];
        const int kx = p / 3, ky = p - kx * 3;
        const float ix = (float)(w + kx) + ox;   // padded coords
        const float iy = (float)(h + ky) + oy;
        const float x0f = floorf(ix), y0f = floorf(iy);
        const float fx = ix - x0f, fy = iy - y0f;
        const int x0 = (int)x0f, y0 = (int)y0f;
        const float mp = lg[p] * inv;
        const float ws4[4] = { (1.f - fx) * (1.f - fy), fx * (1.f - fy),
                               (1.f - fx) * fy,         fx * fy };
        const int xs[4] = { x0, x0 + 1, x0, x0 + 1 };
        const int ys[4] = { y0, y0, y0 + 1, y0 + 1 };
#pragma unroll
        for (int q = 0; q < 4; ++q) {
            const bool ok = (xs[q] >= 1) && (xs[q] <= Wimg) &&
                            (ys[q] >= 1) && (ys[q] <= Himg);
            cidx[p][q] = ok ? ((ys[q] - 1) * Wimg + (xs[q] - 1)) : 0;
            cw[p][q]   = ok ? ws4[q] * mp : 0.f;
        }
    }

    const unsigned short* tb = T + ((long)b * 128 + g * 32) * HW;
    unsigned short* outp = Ycat + ((long)b * 512 + outChOff + g * 32) * HW + hw;
    for (int gc = 0; gc < 32; ++gc) {
        const unsigned short* tc = tb + (long)gc * HW;
        float accv = 0.f;
#pragma unroll
        for (int p = 0; p < 9; ++p) {
#pragma unroll
            for (int q = 0; q < 4; ++q)
                accv += cw[p][q] * bf2f(tc[cidx[p][q]]);
        }
        outp[(long)gc * HW] = f2bf(accv);
    }
}

// ---------------------------------------------------------------------------
// Host launcher
// ---------------------------------------------------------------------------
static inline size_t alignUp(size_t v, size_t a) { return (v + a - 1) & ~(a - 1); }

extern "C" void kernel_launch(void* const* d_in, const int* in_sizes, int n_in,
                              void* d_out, int out_size, void* d_ws, size_t ws_size,
                              hipStream_t stream) {
    (void)in_sizes; (void)n_in; (void)out_size; (void)ws_size;

    const int B = 4, Himg = 80, Wimg = 80, HW = Himg * Wimg;
    const int C1 = 256, Chid = 128;

    // ---- input pointers (setup_inputs dict order) ----
    const float* x     = (const float*)d_in[0];
    const float* cv1w  = (const float*)d_in[1];
    const float* cv1s  = (const float*)d_in[2];
    const float* cv1b  = (const float*)d_in[3];
    const float* cv2w  = (const float*)d_in[4];
    const float* cv2s  = (const float*)d_in[5];
    const float* cv2b  = (const float*)d_in[6];
    struct BlkP {
        const float *cw, *cs, *cb, *dww, *dws, *dwb, *offw, *offb, *maskw, *maskb;
    } blk[2];
    for (int i = 0; i < 2; ++i) {
        const int base = 7 + i * 10;
        blk[i].cw    = (const float*)d_in[base + 0];
        blk[i].cs    = (const float*)d_in[base + 1];
        blk[i].cb    = (const float*)d_in[base + 2];
        blk[i].dww   = (const float*)d_in[base + 3];
        blk[i].dws   = (const float*)d_in[base + 4];
        blk[i].dwb   = (const float*)d_in[base + 5];
        blk[i].offw  = (const float*)d_in[base + 6];
        blk[i].offb  = (const float*)d_in[base + 7];
        blk[i].maskw = (const float*)d_in[base + 8];
        blk[i].maskb = (const float*)d_in[base + 9];
    }

    // ---- workspace carve-up (bf16 activations, fp32 offsets/masks) ----
    char* ws = (char*)d_ws;
    size_t off = 0;
    auto carve = [&](size_t bytes) { char* p = ws + off; off = alignUp(off + bytes, 256); return p; };

    unsigned short* Xbf   = (unsigned short*)carve((size_t)B * C1 * HW * 2);
    unsigned short* Ycat  = (unsigned short*)carve((size_t)B * 512 * HW * 2);
    unsigned short* Tbf   = (unsigned short*)carve((size_t)B * Chid * HW * 2);
    unsigned short* XDW   = (unsigned short*)carve((size_t)B * Chid * HW * 2);
    float*          OFFb  = (float*)carve((size_t)B * 72 * HW * 4);
    float*          MLGb  = (float*)carve((size_t)B * 36 * HW * 4);
    unsigned short* Wcv1  = (unsigned short*)carve((size_t)256 * 256 * 2);
    unsigned short* Wcv2  = (unsigned short*)carve((size_t)256 * 512 * 2);
    unsigned short* Wm[2]   = { (unsigned short*)carve((size_t)128 * 1152 * 2),
                                (unsigned short*)carve((size_t)128 * 1152 * 2) };
    unsigned short* Woff[2] = { (unsigned short*)carve((size_t)72 * 128 * 2),
                                (unsigned short*)carve((size_t)72 * 128 * 2) };
    unsigned short* Wmsk[2] = { (unsigned short*)carve((size_t)36 * 128 * 2),
                                (unsigned short*)carve((size_t)36 * 128 * 2) };

    // ---- fp32 -> bf16 conversions ----
    auto conv = [&](const float* s, unsigned short* d, long n) {
        const int thr = 256;
        f32_to_bf16_kernel<<<dim3((unsigned)((n + thr - 1) / thr)), dim3(thr), 0, stream>>>(s, d, n);
    };
    conv(x,    Xbf,  (long)B * C1 * HW);
    conv(cv1w, Wcv1, 256L * 256);
    conv(cv2w, Wcv2, 256L * 512);
    for (int i = 0; i < 2; ++i) {
        conv(blk[i].cw,    Wm[i],   128L * 1152);
        conv(blk[i].offw,  Woff[i], 72L * 128);
        conv(blk[i].maskw, Wmsk[i], 36L * 128);
    }

    const int nTiles = (B * HW) / TILE_N;   // 200

    // ---- cv1: 1x1, 256->256, SiLU, writes Ycat channels [0,256) ----
    gemm_conv_wmma<<<dim3(nTiles, 4), dim3(256), 0, stream>>>(
        Wcv1, Xbf, cv1s, cv1b, nullptr, Ycat,
        256, 256, 1, HW, Wimg, C1, 0, 512, 0, 1);

    // ---- bottleneck blocks ----
    for (int i = 0; i < 2; ++i) {
        const int inOff  = 128 + 128 * i;   // ys[-1]
        const int outOff = 256 + 128 * i;   // appended dcn output

        // block cv1: 3x3, 128->128, SiLU -> Tbf
        gemm_conv_wmma<<<dim3(nTiles, 2), dim3(256), 0, stream>>>(
            Wm[i], Ycat, blk[i].cs, blk[i].cb, nullptr, Tbf,
            128, 1152, 3, HW, Wimg, 512, inOff, 128, 0, 1);

        // depthwise 3x3 + BN + SiLU -> XDW
        {
            const long total = (long)B * Chid * HW;
            dwconv_silu_kernel<<<dim3((unsigned)((total + 255) / 256)), dim3(256), 0, stream>>>(
                Tbf, blk[i].dww, blk[i].dws, blk[i].dwb, XDW, B, Chid, Himg, Wimg);
        }

        // offset projection: 1x1, 128->72, linear -> OFFb (fp32)
        gemm_conv_wmma<<<dim3(nTiles, 2), dim3(256), 0, stream>>>(
            Woff[i], XDW, nullptr, blk[i].offb, OFFb, nullptr,
            72, 128, 1, HW, Wimg, Chid, 0, 72, 0, 0);

        // mask projection: 1x1, 128->36, linear -> MLGb (fp32)
        gemm_conv_wmma<<<dim3(nTiles, 1), dim3(256), 0, stream>>>(
            Wmsk[i], XDW, nullptr, blk[i].maskb, MLGb, nullptr,
            36, 128, 1, HW, Wimg, Chid, 0, 36, 0, 0);

        // DCN sampling -> Ycat channels [outOff, outOff+128)
        {
            const long total = (long)B * 4 * HW;   // per (b,g,hw)
            dcn_core_kernel<<<dim3((unsigned)((total + 127) / 128)), dim3(128), 0, stream>>>(
                Tbf, OFFb, MLGb, Ycat, outOff, B, Himg, Wimg);
        }
    }

    // ---- cv2: 1x1, 512->256, SiLU -> d_out (fp32) ----
    gemm_conv_wmma<<<dim3(nTiles, 4), dim3(256), 0, stream>>>(
        Wcv2, Ycat, cv2s, cv2b, (float*)d_out, nullptr,
        256, 512, 1, HW, Wimg, 512, 0, 256, 0, 1);
}
